// MTGFlowCNF_35459249996134
// MI455X (gfx1250) — matrix operand order repaired and checked by reading, not compiled, not run on hardware
//
#include <hip/hip_runtime.h>
#include <hip/hip_bf16.h>
#include <math.h>
#include <stdint.h>

// ---------------------------------------------------------------------------
// MTGFlow-CNF for MI455X (gfx1250, wave32, WMMA bf16 16x16x32 w/ f32 accum)
// ---------------------------------------------------------------------------
#define NN   256
#define KK   38
#define TT   100
#define HH   128
#define CTXD 128
#define MM   (NN * KK)        // 9728 rows
#define ZD   100
#define CNFH 512
#define XPAD 256              // padded CNF input width (236 -> 256)
#define STEPS 10

typedef __attribute__((ext_vector_type(16))) __bf16 v16bf;
typedef __attribute__((ext_vector_type(8)))  float  v8f;

__device__ __forceinline__ v8f wmma_bf16(v16bf a, v16bf b, v8f c) {
  return __builtin_amdgcn_wmma_f32_16x16x32_bf16(false, a, false, b,
                                                 (short)0, c, false, false);
}

// A fragment: 16x32 tile, row-major, lda in elements.
// A k order used here (contiguous, matches B): lanes 0-15 -> M=lane, K=0..15;
// lanes 16-31 -> M=lane-16, K=16..31.
template <typename T>
__device__ __forceinline__ v16bf frag_a(const T* A, int lda) {
  int lane = threadIdx.x & 31;
  int m  = lane & 15;
  int hb = (lane >> 4) << 4;          // 0 or 16
  v16bf a;
#pragma unroll
  for (int v = 0; v < 8; ++v) {
    int kb = 2 * v + hb;
    a[2 * v]     = (__bf16)(float)A[m * lda + kb];
    a[2 * v + 1] = (__bf16)(float)A[m * lda + kb + 1];
  }
  return a;
}

// B fragment: 32x16 tile from global f32 matrix with generic strides:
// element(k,n) at B[k*brk + n*brn]; guarded by Kv/Nv (zeros outside).
// B k order used here (contiguous, matches frag_a): lanes 0-15 hold K=0..15, lanes 16-31 hold K=16..31; N=lane&15.
__device__ __forceinline__ v16bf frag_b_f32(const float* B, int brk, int brn,
                                            int k0, int n0, int Kv, int Nv) {
  int lane = threadIdx.x & 31;
  int n  = n0 + (lane & 15);
  int kb = k0 + ((lane >> 4) << 4);
  v16bf b;
#pragma unroll
  for (int e = 0; e < 16; ++e) {
    int k = kb + e;
    float x = (k < Kv && n < Nv) ? B[(size_t)k * brk + (size_t)n * brn] : 0.0f;
    b[e] = (__bf16)x;
  }
  return b;
}

#define V8F_ZERO {0.f,0.f,0.f,0.f,0.f,0.f,0.f,0.f}

// ---------------------------------------------------------------------------
// 0) init: zero dlp/dyn/Cpool, copy x -> z0
// ---------------------------------------------------------------------------
__global__ void init_kernel(const float* __restrict__ x, float* z,
                            float* Cpool, float* dlp, float* dyn) {
  size_t i = (size_t)blockIdx.x * 256 + threadIdx.x;
  if (i < (size_t)MM * CTXD) Cpool[i] = 0.f;
  if (i < (size_t)MM * ZD)   z[i] = x[i];
  if (i < (size_t)MM)        dlp[i] = 0.f;
  if (i == 0)                dyn[0] = 0.f;
}

// ---------------------------------------------------------------------------
// 0b) pre-swizzle a weight matrix into WMMA B-fragment order (bf16), once.
//     out[((kt*NTpad+nt)*32+lane)*16+e] = B(k,n), zero outside Kv/Nv.
// ---------------------------------------------------------------------------
__global__ void swizzleB_kernel(const float* __restrict__ B, int brk, int brn,
                                int Kv, int Nv, int KT, int NTpad,
                                __bf16* out) {
  size_t total = (size_t)KT * NTpad * 512;
  size_t i = (size_t)blockIdx.x * 256 + threadIdx.x;
  if (i >= total) return;
  int e    = (int)(i & 15);
  int lane = (int)((i >> 4) & 31);
  size_t tile = i >> 9;
  int nt = (int)(tile % NTpad);
  int kt = (int)(tile / NTpad);
  int n = nt * 16 + (lane & 15);
  int k = kt * 32 + ((lane >> 4) << 4) + e;
  float v = (k < Kv && n < Nv) ? B[(size_t)k * brk + (size_t)n * brn] : 0.f;
  out[i] = (__bf16)v;
}

// ---------------------------------------------------------------------------
// 1) q/k projections
// ---------------------------------------------------------------------------
__global__ void qk_kernel(const float* __restrict__ x,
                          const float* __restrict__ Wq, const float* __restrict__ bq,
                          const float* __restrict__ Wk, const float* __restrict__ bk,
                          float* q, float* kk) {
  size_t i = (size_t)blockIdx.x * 256 + threadIdx.x;
  if (i >= (size_t)MM * ZD) return;
  int m = (int)(i / ZD), f = (int)(i % ZD);
  const float* xr  = x  + (size_t)m * ZD;
  const float* wqr = Wq + (size_t)f * ZD;
  const float* wkr = Wk + (size_t)f * ZD;
  float sq = bq[f], sk = bk[f];
#pragma unroll 4
  for (int c = 0; c < ZD; ++c) { float xv = xr[c]; sq += xv * wqr[c]; sk += xv * wkr[c]; }
  q[i] = sq; kk[i] = sk;
}

// ---------------------------------------------------------------------------
// 2) attention scores + row softmax -> A[n][k][j]
// ---------------------------------------------------------------------------
__global__ void attn_kernel(const float* __restrict__ q, const float* __restrict__ kk,
                            float* Aadj) {
  int n = blockIdx.x, k = blockIdx.y;
  int j = threadIdx.x;                         // 64 threads
  __shared__ float s[KK];
  __shared__ float mred, sred;
  if (j < KK) {
    const float* qr = q  + (size_t)(n * KK + k) * ZD;
    const float* kr = kk + (size_t)(n * KK + j) * ZD;
    float d = 0.f;
    for (int f = 0; f < ZD; ++f) d += qr[f] * kr[f];
    s[j] = d * 0.1f;
  }
  __syncthreads();
  if (j == 0) {
    float mx = s[0];
    for (int i = 1; i < KK; ++i) mx = fmaxf(mx, s[i]);
    float sum = 0.f;
    for (int i = 0; i < KK; ++i) sum += __expf(s[i] - mx);
    mred = mx; sred = sum;
  }
  __syncthreads();
  if (j < KK)
    Aadj[((size_t)n * KK + k) * KK + j] = __expf(s[j] - mred) / sred;
}

// ---------------------------------------------------------------------------
// 3) LSTM: W_hh^T fragments register-resident across all 100 timesteps.
// ---------------------------------------------------------------------------
__global__ void __launch_bounds__(256)
lstm_kernel(const float* __restrict__ x,  const float* __restrict__ w_ih,
            const float* __restrict__ w_hh, const float* __restrict__ b_ih,
            const float* __restrict__ b_hh, __bf16* hr) {
  __shared__ float  xS[16][TT];
  __shared__ __align__(16) __bf16 hS[16][HH + 8];
  __shared__ float  cS[16][HH];
  __shared__ float  gS[16][4 * HH];
  __shared__ float  wihS[4 * HH], bS[4 * HH];
  int m0  = blockIdx.x * 16;
  int tid = threadIdx.x;
  for (int i = tid; i < 16 * TT; i += 256) {
    int r = i / TT, c = i % TT;
    xS[r][c] = x[(size_t)(m0 + r) * TT + c];
  }
  for (int i = tid; i < 4 * HH; i += 256) { wihS[i] = w_ih[i]; bS[i] = b_ih[i] + b_hh[i]; }
  for (int i = tid; i < 16 * HH; i += 256) {
    int r = i / HH, c = i % HH;
    hS[r][c] = (__bf16)0.f; cS[r][c] = 0.f;
  }
  __syncthreads();

  int wave = tid >> 5, lane = tid & 31;
  int ncol0 = wave * 64;                       // 8 waves x 64 gate cols = 512
  v16bf bf[4][4];
#pragma unroll
  for (int ct = 0; ct < 4; ++ct)
#pragma unroll
    for (int ks = 0; ks < 4; ++ks)
      bf[ct][ks] = frag_b_f32(w_hh, 1, HH, ks * 32, ncol0 + ct * 16, HH, 4 * HH);

  int nl = lane & 15;
  int mh = (lane >> 4) * 8;
  for (int t = 0; t < TT; ++t) {
    v16bf af[4];
#pragma unroll
    for (int ks = 0; ks < 4; ++ks) af[ks] = frag_a(&hS[0][ks * 32], HH + 8);
    v8f acc[4] = {V8F_ZERO, V8F_ZERO, V8F_ZERO, V8F_ZERO};
#pragma unroll
    for (int ct = 0; ct < 4; ++ct)
#pragma unroll
      for (int ks = 0; ks < 4; ++ks)
        acc[ct] = wmma_bf16(af[ks], bf[ct][ks], acc[ct]);
#pragma unroll
    for (int ct = 0; ct < 4; ++ct) {
      int n = ncol0 + ct * 16 + nl;
#pragma unroll
      for (int g2 = 0; g2 < 8; ++g2) {
        int m = mh + g2;
        gS[m][n] = acc[ct][g2] + xS[m][t] * wihS[n] + bS[n];
      }
    }
    __syncthreads();
    for (int i = tid; i < 16 * HH; i += 256) {
      int r = i / HH, c = i % HH;
      float ig = 1.f / (1.f + __expf(-gS[r][c]));
      float fg = 1.f / (1.f + __expf(-gS[r][c + HH]));
      float gg = tanhf(gS[r][c + 2 * HH]);
      float og = 1.f / (1.f + __expf(-gS[r][c + 3 * HH]));
      float cn = fg * cS[r][c] + ig * gg;
      float hn = og * tanhf(cn);
      cS[r][c] = cn;
      hS[r][c] = (__bf16)hn;
      hr[((size_t)(m0 + r) * TT + t) * HH + c] = (__bf16)hn;
    }
    __syncthreads();
  }
}

// ---------------------------------------------------------------------------
// 4) GNN: A-mix + wn GEMM + async-staged shifted residual (wr GEMM), relu,
//    w2 GEMM, masked time-mean -> Cpool.
//    Hp rows (256B each) are staged with GLOBAL_LOAD_ASYNC_TO_LDS_B128
//    (ASYNCcnt) overlapping the VALU A-mix, then s_wait_asynccnt + barrier.
// ---------------------------------------------------------------------------
__global__ void __launch_bounds__(256)
gnn_kernel(const __bf16* __restrict__ hr, const float* __restrict__ Aadj,
           const float* __restrict__ wn, const float* __restrict__ bn,
           const float* __restrict__ wr, const float* __restrict__ w2,
           const float* __restrict__ b2, float* Cpool) {
  __shared__ __align__(16) __bf16 Xm[16][HH + 8];
  __shared__ __align__(16) __bf16 Hp[16][HH + 8];
  __shared__ __align__(16) __bf16 H1[16][HH + 8];
  __shared__ float  Acol[KK];
  int nj = blockIdx.x;
  int n  = nj / KK, j = nj % KK;
  int t0 = blockIdx.y * 16;
  int tid = threadIdx.x;

  // --- async global->LDS stage of Hp rows (shifted residual source) ---
  {
    int r  = tid >> 4;                 // 0..15 row of tile
    int ch = tid & 15;                 // 16B chunk within 256B row
    int t  = t0 + r;
    if (t >= 1 && t <= TT) {
      const __bf16* src = hr + ((size_t)nj * TT + (t - 1)) * HH + ch * 8;
      uint32_t ldso = (uint32_t)(uintptr_t)(&Hp[r][ch * 8]);
      asm volatile("global_load_async_to_lds_b128 %0, %1, off"
                   :: "v"(ldso), "v"(src) : "memory");
    }
  }

  if (tid < KK) Acol[tid] = Aadj[((size_t)n * KK + tid) * KK + j];
  __syncthreads();
  for (int i = tid; i < 16 * HH; i += 256) {
    int r = i >> 7, c = i & (HH - 1);
    int t = t0 + r;
    float s = 0.f;
    if (t < TT) {
#pragma unroll 2
      for (int k2 = 0; k2 < KK; ++k2)
        s += Acol[k2] * (float)hr[((size_t)(n * KK + k2) * TT + t) * HH + c];
    }
    Xm[r][c] = (__bf16)s;
  }
  asm volatile("s_wait_asynccnt 0x0" ::: "memory");
  {                                     // zero rows with no valid t-1 source
    int r = tid >> 4, ch = tid & 15;
    int t = t0 + r;
    if (!(t >= 1 && t <= TT)) {
#pragma unroll
      for (int e = 0; e < 8; ++e) Hp[r][ch * 8 + e] = (__bf16)0.f;
    }
  }
  __syncthreads();

  int wave = tid >> 5, lane = tid & 31;
  int n0 = wave * 16;                          // 8 waves x 16 cols = 128
  int nl = lane & 15, mh = (lane >> 4) * 8;
  v8f acc = V8F_ZERO;
#pragma unroll
  for (int ks = 0; ks < 4; ++ks) {
    acc = wmma_bf16(frag_a(&Xm[0][ks * 32], HH + 8),
                    frag_b_f32(wn, 1, HH, ks * 32, n0, HH, HH), acc);
    acc = wmma_bf16(frag_a(&Hp[0][ks * 32], HH + 8),
                    frag_b_f32(wr, 1, HH, ks * 32, n0, HH, HH), acc);
  }
  float bnv = bn[n0 + nl];
#pragma unroll
  for (int g2 = 0; g2 < 8; ++g2)
    H1[mh + g2][n0 + nl] = (__bf16)fmaxf(acc[g2] + bnv, 0.f);
  __syncthreads();

  v8f acc2 = V8F_ZERO;
#pragma unroll
  for (int ks = 0; ks < 4; ++ks)
    acc2 = wmma_bf16(frag_a(&H1[0][ks * 32], HH + 8),
                     frag_b_f32(w2, 1, HH, ks * 32, n0, HH, HH), acc2);
  float b2v = b2[n0 + nl];
  float part = 0.f;
#pragma unroll
  for (int g2 = 0; g2 < 8; ++g2) {
    int t = t0 + mh + g2;
    if (t < TT) part += acc2[g2] + b2v;
  }
  part *= (1.f / (float)TT);
  part += __shfl_down(part, 16);
  if (lane < 16)
    atomicAdd(&Cpool[(size_t)nj * CTXD + n0 + nl], part);
}

// ---------------------------------------------------------------------------
// 5) ctx projection + layernorm
// ---------------------------------------------------------------------------
__global__ void ctx_kernel(const float* __restrict__ Cpool,
                           const float* __restrict__ Wc, const float* __restrict__ bc,
                           const float* __restrict__ g, const float* __restrict__ b,
                           float* ctx) {
  int row = blockIdx.x, n = threadIdx.x;       // 128 threads
  __shared__ float cp[CTXD], red[CTXD];
  float s = bc[n];
  const float* pr = Cpool + (size_t)row * CTXD;
  const float* wrp = Wc + (size_t)n * CTXD;
  for (int h = 0; h < CTXD; ++h) s += pr[h] * wrp[h];
  cp[n] = s; red[n] = s;
  __syncthreads();
  for (int off = 64; off > 0; off >>= 1) { if (n < off) red[n] += red[n + off]; __syncthreads(); }
  float mu = red[0] * (1.f / CTXD);
  __syncthreads();
  float d = cp[n] - mu;
  red[n] = d * d;
  __syncthreads();
  for (int off = 64; off > 0; off >>= 1) { if (n < off) red[n] += red[n + off]; __syncthreads(); }
  float var = red[0] * (1.f / CTXD);
  ctx[(size_t)row * CTXD + n] = d * rsqrtf(var + 1e-5f) * g[n] + b[n];
}

// ---------------------------------------------------------------------------
// 6) CNF per-step build: X = [z | te(t) | ctx | 0-pad], E = Rademacher noise
// ---------------------------------------------------------------------------
__global__ void cnf_build(const float* __restrict__ z, const float* __restrict__ ctx,
                          const int* __restrict__ en, float t, float* X, float* E) {
  size_t i = (size_t)blockIdx.x * 256 + threadIdx.x;
  if (i >= (size_t)MM * XPAD) return;
  int m = (int)(i >> 8), c = (int)(i & (XPAD - 1));
  float v;
  if (c < ZD)        v = z[(size_t)m * ZD + c];
  else if (c < 108) {
    int f = c - 100;
    float ph = t * 3.14159265358979f * (float)((f & 3) + 1);
    v = (f < 4) ? __sinf(ph) : __cosf(ph);
  }
  else if (c < 236)  v = ctx[(size_t)m * CTXD + (c - 108)];
  else               v = 0.f;
  X[i] = v;
  if (c < 128)
    E[(size_t)m * 128 + c] = (c < ZD) ? (float)(en[(size_t)m * ZD + c] * 2 - 1) : 0.f;
}

// ---------------------------------------------------------------------------
// 7) CNF GEMM: 256-thread block = 16 rows x 128 cols (8 waves, 1 tile each).
//    A panel staged once in LDS (bf16); B fragments are contiguous 32B vector
//    loads from the pre-swizzled bf16 weight buffers.
//    mode 0: +bias; mode 1: tanh(+bias); mode 2: *(1 - mask^2)
// ---------------------------------------------------------------------------
__global__ void __launch_bounds__(256)
gemm_cnf(const float* __restrict__ A, int lda, int kshift,
         const __bf16* __restrict__ Bsw, int NTpad, int Kpad, int Nv,
         const float* __restrict__ bias, int mode,
         const float* __restrict__ mask, int ldm,
         float* C, int ldc) {
  __shared__ __align__(16) __bf16 As[16][CNFH + 8];
  int m0  = blockIdx.x * 16;
  int tid = threadIdx.x;
  for (int i = tid; i < (16 << kshift); i += 256) {
    int r = i >> kshift, c = i & (Kpad - 1);
    As[r][c] = (__bf16)A[(size_t)(m0 + r) * lda + c];
  }
  __syncthreads();

  int wave = tid >> 5, lane = tid & 31;
  int nt = blockIdx.y * 8 + wave;
  v8f acc = V8F_ZERO;
  for (int kt = 0; kt < (Kpad >> 5); ++kt) {
    v16bf a = frag_a(&As[0][kt * 32], CNFH + 8);
    v16bf b = *(const v16bf*)(Bsw + ((((size_t)kt * NTpad + nt) * 32 + lane) << 4));
    acc = wmma_bf16(a, b, acc);
  }
  int nl = nt * 16 + (lane & 15);
  int mh = m0 + (lane >> 4) * 8;
  float bv = (bias && nl < Nv) ? bias[nl] : 0.f;
#pragma unroll
  for (int g2 = 0; g2 < 8; ++g2) {
    int m = mh + g2;
    float v = acc[g2];
    if (mode == 1)      v = tanhf(v + bv);
    else if (mode == 2) { float mm = (nl < Nv) ? mask[(size_t)m * ldm + nl] : 0.f; v *= (1.f - mm * mm); }
    else                v += bv;
    if (nl < Nv) C[(size_t)m * ldc + nl] = v;
  }
}

// ---------------------------------------------------------------------------
// 8) CNF step finalize: tr = <vz,e>, dlp -= tr*dt, z += dz*dt, accum sum(dz^2)
// ---------------------------------------------------------------------------
__global__ void cnf_finalize(const float* __restrict__ vz, const float* __restrict__ E,
                             const float* __restrict__ dz, float* z, float* dlp,
                             float* dyn_acc, float dt) {
  int m = blockIdx.x * 64 + threadIdx.x;
  float tr = 0.f, dsq = 0.f;
  for (int c = 0; c < ZD; ++c) {
    float e = E[(size_t)m * 128 + c];
    tr += vz[(size_t)m * 128 + c] * e;
    float d = dz[(size_t)m * 128 + c];
    dsq += d * d;
    z[(size_t)m * ZD + c] += d * dt;
  }
  dlp[m] -= tr * dt;
  __shared__ float red[64];
  red[threadIdx.x] = dsq;
  __syncthreads();
  for (int off = 32; off > 0; off >>= 1) {
    if (threadIdx.x < off) red[threadIdx.x] += red[threadIdx.x + off];
    __syncthreads();
  }
  if (threadIdx.x == 0) atomicAdd(dyn_acc, red[0]);
}

// ---------------------------------------------------------------------------
// 9) final: log_px = log_pz(z1; mu[m%K]) - dlp;  dyn_norm at out[MM]
// ---------------------------------------------------------------------------
__global__ void final_kernel(const float* __restrict__ z, const float* __restrict__ mu,
                             const float* __restrict__ dlp, const float* __restrict__ dyn,
                             float* out) {
  int m = blockIdx.x * 128 + threadIdx.x;
  if (m < MM) {
    const float* muv = mu + (size_t)(m % KK) * ZD;
    float s = 0.f;
    for (int c = 0; c < ZD; ++c) { float d = z[(size_t)m * ZD + c] - muv[c]; s += d * d; }
    float logpz = -0.5f * s - 0.5f * (float)ZD * logf(2.f * 3.14159265358979f);
    out[m] = logpz - dlp[m];
  }
  if (blockIdx.x == 0 && threadIdx.x == 0)
    out[MM] = dyn[0] / ((float)STEPS * (float)MM * (float)ZD);
}

// ---------------------------------------------------------------------------
// Host launcher
// ---------------------------------------------------------------------------
extern "C" void kernel_launch(void* const* d_in, const int* in_sizes, int n_in,
                              void* d_out, int out_size, void* d_ws, size_t ws_size,
                              hipStream_t stream) {
  (void)in_sizes; (void)n_in; (void)out_size; (void)ws_size;
  const float* x     = (const float*)d_in[0];
  const float* wqW   = (const float*)d_in[1];
  const float* wqB   = (const float*)d_in[2];
  const float* wkW   = (const float*)d_in[3];
  const float* wkB   = (const float*)d_in[4];
  const float* w_ih  = (const float*)d_in[5];
  const float* w_hh  = (const float*)d_in[6];
  const float* b_ih  = (const float*)d_in[7];
  const float* b_hh  = (const float*)d_in[8];
  const float* wnW   = (const float*)d_in[9];
  const float* wnB   = (const float*)d_in[10];
  const float* wrW   = (const float*)d_in[11];
  const float* w2W   = (const float*)d_in[12];
  const float* w2B   = (const float*)d_in[13];
  const float* wcW   = (const float*)d_in[14];
  const float* wcB   = (const float*)d_in[15];
  const float* lnG   = (const float*)d_in[16];
  const float* lnB   = (const float*)d_in[17];
  const float* W1    = (const float*)d_in[18];
  const float* B1    = (const float*)d_in[19];
  const float* W2    = (const float*)d_in[20];
  const float* B2    = (const float*)d_in[21];
  const float* W3    = (const float*)d_in[22];
  const float* B3    = (const float*)d_in[23];
  const float* W4    = (const float*)d_in[24];
  const float* B4    = (const float*)d_in[25];
  const float* mu    = (const float*)d_in[26];
  const int*   enoi  = (const int*)d_in[27];
  float* out = (float*)d_out;

  char* ws = (char*)d_ws;
  size_t off = 0;
  auto take = [&](size_t bytes) -> char* {
    off = (off + 255) & ~(size_t)255;
    char* p = ws + off;
    off += bytes;
    return p;
  };
  float*  q     = (float*)take((size_t)MM * ZD * 4);
  float*  kbuf  = (float*)take((size_t)MM * ZD * 4);
  float*  Aadj  = (float*)take((size_t)NN * KK * KK * 4);
  __bf16* hr    = (__bf16*)take((size_t)MM * TT * HH * 2);    // 249 MB
  float*  Cpool = (float*)take((size_t)MM * CTXD * 4);
  float*  ctx   = (float*)take((size_t)MM * CTXD * 4);
  float*  z     = (float*)take((size_t)MM * ZD * 4);
  float*  X     = (float*)take((size_t)MM * XPAD * 4);
  float*  E     = (float*)take((size_t)MM * 128 * 4);
  float*  h1    = (float*)take((size_t)MM * CNFH * 4);
  float*  h2    = (float*)take((size_t)MM * CNFH * 4);
  float*  h3    = (float*)take((size_t)MM * CNFH * 4);
  float*  dz    = (float*)take((size_t)MM * 128 * 4);
  float*  u3    = (float*)take((size_t)MM * CNFH * 4);
  float*  u2    = (float*)take((size_t)MM * CNFH * 4);
  float*  u1    = (float*)take((size_t)MM * CNFH * 4);
  float*  vz    = (float*)take((size_t)MM * 128 * 4);
  float*  dlp   = (float*)take((size_t)MM * 4);
  float*  dyn   = (float*)take(4);
  // pre-swizzled bf16 B buffers (fragment order): {KT, NTpad}
  __bf16* B1T = (__bf16*)take((size_t)8  * 32 * 512 * 2);   // fwd1: W1^T
  __bf16* B2T = (__bf16*)take((size_t)16 * 32 * 512 * 2);   // fwd2: W2^T
  __bf16* B3T = (__bf16*)take((size_t)16 * 32 * 512 * 2);   // fwd3: W3^T
  __bf16* B4T = (__bf16*)take((size_t)16 * 8  * 512 * 2);   // fwd4: W4^T
  __bf16* B4b = (__bf16*)take((size_t)4  * 32 * 512 * 2);   // bwd1: W4
  __bf16* B3b = (__bf16*)take((size_t)16 * 32 * 512 * 2);   // bwd2: W3
  __bf16* B2b = (__bf16*)take((size_t)16 * 32 * 512 * 2);   // bwd3: W2
  __bf16* B1b = (__bf16*)take((size_t)16 * 8  * 512 * 2);   // bwd4: W1

  const int MT = MM / 16;                      // 608 row tiles
  auto swz = [&](const float* src, int brk, int brn, int Kv, int Nv,
                 int KT, int NTpad, __bf16* dst) {
    size_t total = (size_t)KT * NTpad * 512;
    swizzleB_kernel<<<(int)((total + 255) / 256), 256, 0, stream>>>(
        src, brk, brn, Kv, Nv, KT, NTpad, dst);
  };

  init_kernel<<<(MM * CTXD + 255) / 256, 256, 0, stream>>>(x, z, Cpool, dlp, dyn);
  // pre-swizzle CNF weights (reused across all 10 steps)
  swz(W1, 1, 236, 236, CNFH, 8, 32, B1T);      // B(k,n)=W1[n][k]
  swz(W2, 1, 512, 512, CNFH, 16, 32, B2T);
  swz(W3, 1, 512, 512, CNFH, 16, 32, B3T);
  swz(W4, 1, 512, 512, ZD,   16, 8,  B4T);     // B(k,n)=W4[n][k], n<100
  swz(W4, 512, 1, ZD, CNFH,  4, 32,  B4b);     // B(k,n)=W4[k][n], k<100
  swz(W3, 512, 1, 512, CNFH, 16, 32, B3b);
  swz(W2, 512, 1, 512, CNFH, 16, 32, B2b);
  swz(W1, 236, 1, 512, 236,  16, 8,  B1b);     // B(k,n)=W1[k][n], n<236

  qk_kernel<<<(MM * ZD + 255) / 256, 256, 0, stream>>>(x, wqW, wqB, wkW, wkB, q, kbuf);
  attn_kernel<<<dim3(NN, KK), 64, 0, stream>>>(q, kbuf, Aadj);
  lstm_kernel<<<MT, 256, 0, stream>>>(x, w_ih, w_hh, b_ih, b_hh, hr);
  gnn_kernel<<<dim3(MM, (TT + 15) / 16), 256, 0, stream>>>(hr, Aadj, wnW, wnB, wrW,
                                                           w2W, w2B, Cpool);
  ctx_kernel<<<MM, CTXD, 0, stream>>>(Cpool, wcW, wcB, lnG, lnB, ctx);

  const float dt = 1.0f / STEPS;
  for (int s = 0; s < STEPS; ++s) {
    float t = dt * (float)s;
    cnf_build<<<(MM * XPAD + 255) / 256, 256, 0, stream>>>(
        z, ctx, enoi + (size_t)s * MM * ZD, t, X, E);
    // forward
    gemm_cnf<<<dim3(MT, 4), 256, 0, stream>>>(X,  XPAD, 8, B1T, 32, XPAD, CNFH,
                                              B1, 1, nullptr, 0, h1, CNFH);
    gemm_cnf<<<dim3(MT, 4), 256, 0, stream>>>(h1, CNFH, 9, B2T, 32, CNFH, CNFH,
                                              B2, 1, nullptr, 0, h2, CNFH);
    gemm_cnf<<<dim3(MT, 4), 256, 0, stream>>>(h2, CNFH, 9, B3T, 32, CNFH, CNFH,
                                              B3, 1, nullptr, 0, h3, CNFH);
    gemm_cnf<<<dim3(MT, 1), 256, 0, stream>>>(h3, CNFH, 9, B4T, 8, CNFH, ZD,
                                              B4, 0, nullptr, 0, dz, 128);
    // VJP
    gemm_cnf<<<dim3(MT, 4), 256, 0, stream>>>(E,  128, 7, B4b, 32, 128, CNFH,
                                              nullptr, 2, h3, CNFH, u3, CNFH);
    gemm_cnf<<<dim3(MT, 4), 256, 0, stream>>>(u3, CNFH, 9, B3b, 32, CNFH, CNFH,
                                              nullptr, 2, h2, CNFH, u2, CNFH);
    gemm_cnf<<<dim3(MT, 4), 256, 0, stream>>>(u2, CNFH, 9, B2b, 32, CNFH, CNFH,
                                              nullptr, 2, h1, CNFH, u1, CNFH);
    gemm_cnf<<<dim3(MT, 1), 256, 0, stream>>>(u1, CNFH, 9, B1b, 8, CNFH, 236,
                                              nullptr, 0, nullptr, 0, vz, 128);
    cnf_finalize<<<MM / 64, 64, 0, stream>>>(vz, E, dz, z, dlp, dyn, dt);
  }
  final_kernel<<<(MM + 127) / 128, 128, 0, stream>>>(z, mu, dlp, dyn, out);
}
